// ResidualAttentionBlock_13125420057062
// MI455X (gfx1250) — matrix operand order repaired and checked by reading, not compiled, not run on hardware
//
#include <hip/hip_runtime.h>

// ---------------- problem constants ----------------
#define B_    16
#define FIN_  256
#define N_    1024          // H*W = 32*32
#define NH_   8
#define FH_   64
#define CIN_  258           // FIN + 2 pos channels
#define CPAD_ 288           // padded to multiple of 32 for K-dim of WMMA
#define COUT_ 512

// trivially-constructible vector types (HIP's uint4 has ctors -> breaks unions)
typedef __attribute__((ext_vector_type(4)))  unsigned int u32x4;
typedef __attribute__((ext_vector_type(16))) __bf16       v16bf;
typedef __attribute__((ext_vector_type(8)))  float        v8f;

union Frag {
  v16bf v;
  u32x4 q[2];
  unsigned short u[16];
};

__device__ __forceinline__ unsigned short f2bf(float f) {
  unsigned int u = __builtin_bit_cast(unsigned int, f);
  u += 0x7FFFu + ((u >> 16) & 1u);        // round-to-nearest-even
  return (unsigned short)(u >> 16);
}

// A-fragment (16xK=32, bf16), per ISA 7.12.2:
// lane L (half=L/16, r=L%16): u[0..7] = M[r][k0 + half*8 + i], u[8..15] = M[r][k0+16+half*8+i]
__device__ __forceinline__ Frag load_a(const unsigned short* M, int ld, int row0, int k0) {
  int lane = threadIdx.x & 31;
  int half = lane >> 4;
  int r    = row0 + (lane & 15);
  Frag f;
  f.q[0] = *(const u32x4*)(M + (size_t)r * ld + k0 + half * 8);
  f.q[1] = *(const u32x4*)(M + (size_t)r * ld + k0 + 16 + half * 8);
  return f;
}

// B-fragment (K=32 x 16, bf16): lane L holds column c=col0+L%16, u[0..7] = k0+half*8+i, u[8..15] = k0+16+half*8+i
// source stored "column-major": M[col][k] contiguous in k.
__device__ __forceinline__ Frag load_b(const unsigned short* M, int ld, int col0, int k0) {
  int lane = threadIdx.x & 31;
  int half = lane >> 4;
  int c    = col0 + (lane & 15);
  const u32x4* p = (const u32x4*)(M + (size_t)c * ld + k0 + half * 8);
  Frag f;
  f.q[0] = p[0];
  f.q[1] = p[2];
  return f;
}

__device__ __forceinline__ v8f wmma_bf16(const Frag& a, const Frag& b, v8f c) {
  return __builtin_amdgcn_wmma_f32_16x16x32_bf16(false, a.v, false, b.v,
                                                 (short)0, c, false, false);
}

// ---------------- CDNA5 async load-to-LDS helpers ----------------
#define AS1_ __attribute__((address_space(1)))
#define AS3_ __attribute__((address_space(3)))

// builtin signature (from hipcc diagnostic): (v4i AS1*, v4i AS3*, imm int, imm int)
typedef int vsi4 __attribute__((vector_size(16)));

__device__ __forceinline__ void async_b128(const void* g, void* l) {
#if __has_builtin(__builtin_amdgcn_global_load_async_to_lds_b128)
  AS1_ vsi4* gp = (AS1_ vsi4*)(unsigned long long)(size_t)g;  // AS1 value == generic value
  // generic LDS pointer low 32 bits are the LDS offset (aperture rule, ISA 10.2)
  AS3_ vsi4* lp = (AS3_ vsi4*)(unsigned)(unsigned long long)(size_t)l;
  __builtin_amdgcn_global_load_async_to_lds_b128(gp, lp, 0, 0);
#else
  unsigned lo = (unsigned)(unsigned long long)(size_t)l;   // LDS byte offset
  __asm__ volatile("global_load_async_to_lds_b128 %0, %1, off"
                   :: "v"(lo), "v"(g) : "memory");
#endif
}

__device__ __forceinline__ void wait_async0() {
#if __has_builtin(__builtin_amdgcn_s_wait_asynccnt)
  __builtin_amdgcn_s_wait_asynccnt(0);
#else
  __asm__ volatile("s_wait_asynccnt 0" ::: "memory");
#endif
}

__device__ __forceinline__ void wait_tensor0() {
#if __has_builtin(__builtin_amdgcn_s_wait_tensorcnt)
  __builtin_amdgcn_s_wait_tensorcnt(0);
#else
  __asm__ volatile("s_wait_tensorcnt 0" ::: "memory");
#endif
}

// ---------------- Tensor Data Mover: 2D tile -> LDS (ISA ch.8 D# layout) ----------------
// this toolchain's builtin (from hipcc diagnostic): 6 args
//   (uint32x4 g0, int32x8 g1, int32x4 g2, int32x4 g3, int32x8 g4, i32 cpol)
typedef unsigned int vu4g __attribute__((vector_size(16)));
typedef int          vi8g __attribute__((vector_size(32)));
typedef int          vi4g __attribute__((vector_size(16)));

__device__ __forceinline__ void tdm_load_2d(const void* gsrc, void* ldst,
                                            unsigned tile_d0, unsigned tile_d1,
                                            unsigned tens_d0, unsigned tens_d1,
                                            unsigned stride0 /* elements */) {
#if __has_builtin(__builtin_amdgcn_tensor_load_to_lds)
  unsigned long long ga = (unsigned long long)(size_t)gsrc;
  unsigned           la = (unsigned)(unsigned long long)(size_t)ldst;  // LDS byte offset
  vu4g g0;
  g0[0] = 1u;                                                   // count=1 valid descriptor
  g0[1] = la;                                                   // lds_addr [63:32]
  g0[2] = (unsigned)ga;                                         // global_addr [95:64]
  g0[3] = (unsigned)((ga >> 32) & 0x01FFFFFFu) | (2u << 30);    // addr[56:32] | type=2
  vi8g g1;
  g1[0] = (int)(1u << 16);                                      // data_size=1 (2B), mask=0
  g1[1] = (int)((tens_d0 & 0xFFFFu) << 16);                     // tensor_dim0 lo16 @[63:48]
  g1[2] = (int)(((tens_d0 >> 16) & 0xFFFFu) |
                ((tens_d1 & 0xFFFFu) << 16));                   // dim0 hi | dim1 lo
  g1[3] = (int)(((tens_d1 >> 16) & 0xFFFFu) |
                ((tile_d0 & 0xFFFFu) << 16));                   // dim1 hi | tile_dim0
  g1[4] = (int)(tile_d1 & 0xFFFFu);                             // tile_dim1 (tile_dim2=0)
  g1[5] = (int)stride0;                                         // tensor_dim0_stride[31:0]
  g1[6] = 0;                                                    // stride0 hi | stride1 lo
  g1[7] = 0;
  vi4g z4 = {0, 0, 0, 0};
  vi8g z8 = {0, 0, 0, 0, 0, 0, 0, 0};
  __builtin_amdgcn_tensor_load_to_lds(g0, g1, z4, z4, z8, 0);
#else
  (void)gsrc; (void)ldst; (void)tile_d0; (void)tile_d1;
  (void)tens_d0; (void)tens_d1; (void)stride0;
#endif
}

// ---------------- kernel 1: pack x + pos grid -> T[b][n][CPAD] bf16 ----------------
__global__ __launch_bounds__(256) void pack_t_kernel(const float* __restrict__ x,
                                                     unsigned short* __restrict__ T) {
  int id = blockIdx.x * 256 + threadIdx.x;
  if (id >= B_ * N_ * CPAD_) return;
  int c = id % CPAD_;
  int n = (id / CPAD_) % N_;
  int b = id / (CPAD_ * N_);
  float v = 0.f;
  if (c < FIN_)            v = x[((size_t)b * FIN_ + c) * N_ + n];
  else if (c == FIN_)      v = -1.f + 2.f * (float)(n >> 5) / 31.f;   // gy (row)
  else if (c == FIN_ + 1)  v = -1.f + 2.f * (float)(n & 31) / 31.f;   // gx (col)
  T[id] = f2bf(v);
}

// ---------------- kernel 2: pack weights to bf16 (K zero-padded) ----------------
__global__ __launch_bounds__(256) void pack_w_kernel(const float* __restrict__ wq,
                                                     const float* __restrict__ wk,
                                                     const float* __restrict__ wv,
                                                     const float* __restrict__ wo,
                                                     unsigned short* __restrict__ Wqkv,
                                                     unsigned short* __restrict__ Wo) {
  int id = blockIdx.x * 256 + threadIdx.x;
  const int nqkv = 3 * COUT_ * CPAD_;
  if (id < nqkv) {
    int c = id % CPAD_;
    int o = (id / CPAD_) % COUT_;
    int s = id / (CPAD_ * COUT_);
    const float* w = (s == 0) ? wq : (s == 1) ? wk : wv;
    Wqkv[id] = f2bf(c < CIN_ ? w[(size_t)o * CIN_ + c] : 0.f);
  } else {
    int j = id - nqkv;
    if (j < FIN_ * COUT_) Wo[j] = f2bf(wo[j]);
  }
}

// ---------------- kernel 3: QKV projection GEMM ----------------
// per wave: 16 out-channels x 64 tokens, K-loop over 288 in steps of 32.
// Q,K stored token-major [b][h][n][64]; V stored d-major [b][h][d][1024].
__global__ __launch_bounds__(256) void qkv_kernel(const unsigned short* __restrict__ T,
                                                  const unsigned short* __restrict__ Wqkv,
                                                  const float* __restrict__ bq,
                                                  const float* __restrict__ bk,
                                                  const float* __restrict__ bv,
                                                  unsigned short* __restrict__ Q,
                                                  unsigned short* __restrict__ K,
                                                  unsigned short* __restrict__ V) {
  int w    = blockIdx.x * 8 + (threadIdx.x >> 5);
  int lane = threadIdx.x & 31, half = lane >> 4, l15 = lane & 15;
  int nt = w & 15;  w >>= 4;        // 16 strips of 64 tokens
  int mt = w & 31;  w >>= 5;        // 32 tiles of 16 out-channels
  int s  = w % 3;                   // q/k/v
  int b  = w / 3;
  int m0 = mt * 16, n0 = nt * 64;

  const unsigned short* Wp = Wqkv + (size_t)s * COUT_ * CPAD_;
  const unsigned short* Tb = T + (size_t)b * N_ * CPAD_;

  v8f acc[4] = {};
  for (int kb = 0; kb < CPAD_; kb += 32) {
    // preload all fragments for this K-step so waits retire progressively
    Frag a  = load_a(Wp, CPAD_, m0, kb);
    Frag b0 = load_b(Tb, CPAD_, n0 +  0, kb);
    Frag b1 = load_b(Tb, CPAD_, n0 + 16, kb);
    Frag b2 = load_b(Tb, CPAD_, n0 + 32, kb);
    Frag b3 = load_b(Tb, CPAD_, n0 + 48, kb);
    acc[0] = wmma_bf16(a, b0, acc[0]);
    acc[1] = wmma_bf16(a, b1, acc[1]);
    acc[2] = wmma_bf16(a, b2, acc[2]);
    acc[3] = wmma_bf16(a, b3, acc[3]);
  }

  const float* bias = (s == 0) ? bq : (s == 1) ? bk : bv;
  int h     = m0 >> 6;
  int dbase = (m0 & 63) + half * 8;

  if (s < 2) {
    unsigned short* dst = ((s == 0) ? Q : K) + (size_t)(b * NH_ + h) * N_ * FH_;
    for (int t = 0; t < 4; ++t) {
      int n = n0 + t * 16 + l15;
      unsigned short tmp[8];
      for (int j = 0; j < 8; ++j) tmp[j] = f2bf(acc[t][j] + bias[m0 + half * 8 + j]);
      *(u32x4*)(dst + (size_t)n * FH_ + dbase) = *(u32x4*)tmp;
    }
  } else {
    unsigned short* dst = V + (size_t)(b * NH_ + h) * FH_ * N_;
    for (int t = 0; t < 4; ++t) {
      int n = n0 + t * 16 + l15;
      for (int j = 0; j < 8; ++j)
        dst[(size_t)(dbase + j) * N_ + n] = f2bf(acc[t][j] + bias[m0 + half * 8 + j]);
    }
  }
}

// ---------------- kernel 4: flash attention ----------------
// block = 8 waves handling one (b,h) and 128 q rows; each wave owns 16 q rows.
// K tile (32x64) staged by the Tensor Data Mover (TENSORcnt, one issue per block),
// V tile (64x32) staged by GLOBAL_LOAD_ASYNC_TO_LDS_B128 (ASYNCcnt), both
// double-buffered so DMA overlaps WMMA on the previous block.
__global__ __launch_bounds__(256) void attn_kernel(const unsigned short* __restrict__ Q,
                                                   const unsigned short* __restrict__ K,
                                                   const unsigned short* __restrict__ V,
                                                   unsigned short* __restrict__ O) {
  __shared__ __align__(16) unsigned short Ks[2][32 * 64];      // [buf][key][d]
  __shared__ __align__(16) unsigned short Vs[2][64 * 32];      // [buf][d][key]
  __shared__ __align__(16) unsigned short Ps[8 * 16 * 32];     // per-wave P tile [q][key]

  int bid = blockIdx.x;
  int qb  = bid & 7;
  int h   = (bid >> 3) & 7;
  int b   = bid >> 6;
  int wv_  = threadIdx.x >> 5;
  int lane = threadIdx.x & 31, half = lane >> 4, l15 = lane & 15;
  int tid  = threadIdx.x;
  int d_   = tid >> 2, part_ = tid & 3;

  size_t head = (size_t)(b * NH_ + h) * N_ * FH_;
  const unsigned short* Qh = Q + head;
  const unsigned short* Kh = K + head;
  const unsigned short* Vh = V + head;   // [d][n]

  int q0 = qb * 128 + wv_ * 16;
  Frag aq0 = load_a(Qh, FH_, q0, 0);
  Frag aq1 = load_a(Qh, FH_, q0, 32);

  v8f acc[4] = {};
  float mrow[8], lrow[8];
  for (int j = 0; j < 8; ++j) { mrow[j] = -1e30f; lrow[j] = 0.f; }

  // prologue: stage first K block via TDM (wave 0), first V block via async loads
  if (wv_ == 0)
    tdm_load_2d(Kh, Ks[0], /*tile*/ FH_, 32, /*tensor*/ FH_, N_, /*stride0*/ FH_);
  async_b128(Vh + (size_t)d_ * N_ + part_ * 8, &Vs[0][d_ * 32 + part_ * 8]);
  wait_async0();
  if (wv_ == 0) wait_tensor0();
  __syncthreads();

  int buf = 0;
  for (int kb = 0; kb < N_; kb += 32) {
    // DMA-prefetch next block into the other buffer while we compute
    if (kb + 32 < N_) {
      if (wv_ == 0)
        tdm_load_2d(Kh + (size_t)(kb + 32) * FH_, Ks[buf ^ 1],
                    FH_, 32, FH_, N_, FH_);
      async_b128(Vh + (size_t)d_ * N_ + (kb + 32) + part_ * 8,
                 &Vs[buf ^ 1][d_ * 32 + part_ * 8]);
    }

    // S = Q K^T / sqrt(64): two 16x16 tiles, each K-dim 64 = 2 WMMA
    v8f s0 = {}, s1 = {};
    s0 = wmma_bf16(aq0, load_b(Ks[buf], FH_, 0, 0), s0);
    s0 = wmma_bf16(aq1, load_b(Ks[buf], FH_, 0, 32), s0);
    s1 = wmma_bf16(aq0, load_b(Ks[buf], FH_, 16, 0), s1);
    s1 = wmma_bf16(aq1, load_b(Ks[buf], FH_, 16, 32), s1);

    // online softmax (row r = half*8+j lives in 16 lanes of one half)
    float p0[8], p1[8], sc[8];
    for (int j = 0; j < 8; ++j) {
      float a = s0[j] * 0.125f;
      float c = s1[j] * 0.125f;
      float tmax = fmaxf(a, c);
      for (int msk = 1; msk < 16; msk <<= 1) tmax = fmaxf(tmax, __shfl_xor(tmax, msk, 32));
      float mn = fmaxf(mrow[j], tmax);
      sc[j] = __expf(mrow[j] - mn);
      p0[j] = __expf(a - mn);
      p1[j] = __expf(c - mn);
      float r = p0[j] + p1[j];
      for (int msk = 1; msk < 16; msk <<= 1) r += __shfl_xor(r, msk, 32);
      lrow[j] = lrow[j] * sc[j] + r;
      mrow[j] = mn;
    }
    for (int t = 0; t < 4; ++t)
      for (int j = 0; j < 8; ++j) acc[t][j] *= sc[j];

    // P tile -> per-wave LDS region, then feed back as A-fragment
    unsigned short* Pw = Ps + wv_ * 512;
    for (int j = 0; j < 8; ++j) {
      Pw[(half * 8 + j) * 32 + l15]      = f2bf(p0[j]);
      Pw[(half * 8 + j) * 32 + 16 + l15] = f2bf(p1[j]);
    }
    __asm__ volatile("s_wait_dscnt 0" ::: "memory");  // own-wave LDS RAW

    Frag ap = load_a(Pw, 32, 0, 0);
    for (int t = 0; t < 4; ++t)
      acc[t] = wmma_bf16(ap, load_b(Vs[buf], 32, t * 16, 0), acc[t]);

    wait_async0();                      // next V tile resident
    if (wv_ == 0) wait_tensor0();       // next K tile resident
    __syncthreads();                    // everyone done with current buffer
    buf ^= 1;
  }

  // epilogue: normalize, store O token-major [b][n][512] for the output GEMM
  unsigned short* Ob = O + (size_t)b * N_ * COUT_;
  for (int j = 0; j < 8; ++j) {
    float inv = 1.f / lrow[j];
    int n = q0 + half * 8 + j;
    for (int t = 0; t < 4; ++t)
      Ob[(size_t)n * COUT_ + h * FH_ + t * 16 + l15] = f2bf(acc[t][j] * inv);
  }
}

// ---------------- kernel 5: output projection + bias + residual ----------------
__global__ __launch_bounds__(256) void oproj_kernel(const unsigned short* __restrict__ O,
                                                    const unsigned short* __restrict__ Wo,
                                                    const float* __restrict__ bo,
                                                    const float* __restrict__ x,
                                                    float* __restrict__ out) {
  int w    = blockIdx.x * 8 + (threadIdx.x >> 5);
  int lane = threadIdx.x & 31, half = lane >> 4, l15 = lane & 15;
  int nt = w & 15;  w >>= 4;      // 16 strips of 64 tokens
  int mt = w & 15;  w >>= 4;      // 16 tiles of 16 out-channels
  int b  = w;
  int m0 = mt * 16, n0 = nt * 64;

  const unsigned short* Ob = O + (size_t)b * N_ * COUT_;
  v8f acc[4] = {};
  for (int kc = 0; kc < COUT_; kc += 32) {
    Frag a  = load_a(Wo, COUT_, m0, kc);
    Frag b0 = load_b(Ob, COUT_, n0 +  0, kc);
    Frag b1 = load_b(Ob, COUT_, n0 + 16, kc);
    Frag b2 = load_b(Ob, COUT_, n0 + 32, kc);
    Frag b3 = load_b(Ob, COUT_, n0 + 48, kc);
    acc[0] = wmma_bf16(a, b0, acc[0]);
    acc[1] = wmma_bf16(a, b1, acc[1]);
    acc[2] = wmma_bf16(a, b2, acc[2]);
    acc[3] = wmma_bf16(a, b3, acc[3]);
  }
  for (int t = 0; t < 4; ++t) {
    int n = n0 + t * 16 + l15;
    for (int j = 0; j < 8; ++j) {
      int m = m0 + half * 8 + j;
      size_t idx = ((size_t)b * FIN_ + m) * N_ + n;
      out[idx] = acc[t][j] + bo[m] + x[idx];
    }
  }
}

// ---------------- launch ----------------
extern "C" void kernel_launch(void* const* d_in, const int* in_sizes, int n_in,
                              void* d_out, int out_size, void* d_ws, size_t ws_size,
                              hipStream_t stream) {
  const float* x  = (const float*)d_in[0];
  const float* wq = (const float*)d_in[1];
  const float* bq = (const float*)d_in[2];
  const float* wk = (const float*)d_in[3];
  const float* bk = (const float*)d_in[4];
  const float* wv = (const float*)d_in[5];
  const float* bv = (const float*)d_in[6];
  const float* wo = (const float*)d_in[7];
  const float* bo = (const float*)d_in[8];
  float* out = (float*)d_out;

  char* ws = (char*)d_ws;
  size_t off = 0;
  unsigned short* T    = (unsigned short*)(ws + off); off += (size_t)B_ * N_ * CPAD_ * 2;
  unsigned short* Wqkv = (unsigned short*)(ws + off); off += (size_t)3 * COUT_ * CPAD_ * 2;
  unsigned short* Wo2  = (unsigned short*)(ws + off); off += (size_t)FIN_ * COUT_ * 2;
  unsigned short* Qb   = (unsigned short*)(ws + off); off += (size_t)B_ * NH_ * N_ * FH_ * 2;
  unsigned short* Kb   = (unsigned short*)(ws + off); off += (size_t)B_ * NH_ * N_ * FH_ * 2;
  unsigned short* Vb   = (unsigned short*)(ws + off); off += (size_t)B_ * NH_ * N_ * FH_ * 2;
  unsigned short* Ob   = (unsigned short*)(ws + off); off += (size_t)B_ * N_ * COUT_ * 2;
  (void)ws_size; (void)in_sizes; (void)n_in; (void)out_size;

  pack_t_kernel<<<(B_ * N_ * CPAD_ + 255) / 256, 256, 0, stream>>>(x, T);
  pack_w_kernel<<<((3 * COUT_ * CPAD_ + FIN_ * COUT_) + 255) / 256, 256, 0, stream>>>(
      wq, wk, wv, wo, Wqkv, Wo2);
  qkv_kernel<<<3072, 256, 0, stream>>>(T, Wqkv, bq, bk, bv, Qb, Kb, Vb);
  attn_kernel<<<1024, 256, 0, stream>>>(Qb, Kb, Vb, Ob);
  oproj_kernel<<<512, 256, 0, stream>>>(Ob, Wo2, bo, x, out);
}
